// Interaction_Block_515396076340
// MI455X (gfx1250) — matrix operand order repaired and checked by reading, not compiled, not run on hardware
//
// CDNA5 / gfx1250 implementation of the GNN interaction block.
// Strategy: all dense projections (gated MLPs, FFNs, attention logits) run as
// bf16 WMMA (v_wmma_f32_16x16x32_bf16) with f32 accumulation; weights are
// re-packed once per launch into the wave32 B-fragment layout so every B load
// is a contiguous 32B/lane global_load_b128 pair. Activations are staged
// through LDS (ds_read_b128 A-fragments). Scatter/softmax phases use f32
// global atomics. Memory-bound by roofline (~3-5GB @ 23.3TB/s dominates
// ~2e11 bf16 WMMA FLOPs).

#include <hip/hip_runtime.h>
#include <cstdint>
#include <cstddef>

#define NA_N 5000
#define NB_N 80000
#define ED_N 160000
#define NL_N 120000

typedef __bf16 bf16;
typedef __attribute__((ext_vector_type(16))) __bf16 v16bf;
typedef __attribute__((ext_vector_type(8)))  __bf16 v8bf;
typedef __attribute__((ext_vector_type(8)))  float  v8f;

// ---------------------------------------------------------------- helpers --

__device__ __forceinline__ float siluf(float x) { return x / (1.f + __expf(-x)); }
__device__ __forceinline__ float sigmf(float x) { return 1.f / (1.f + __expf(-x)); }

__device__ __forceinline__ void atomicMaxF32(float* addr, float v) {
    unsigned int* ua = (unsigned int*)addr;
    unsigned int cur = __float_as_uint(*addr);
    while (__uint_as_float(cur) < v) {
        unsigned int prev = atomicCAS(ua, cur, __float_as_uint(v));
        if (prev == cur) break;
        cur = prev;
    }
}

// A fragment (16x32 bf16, MxK) from row-major LDS tile.
// lane L: row m=L&15, K chunks [kb,kb+8) and [16+kb,16+kb+8), kb=(L>>4)*8.
__device__ __forceinline__ v16bf lds_a_frag(const bf16* A, int lda, int kt, int lane) {
    const int m  = lane & 15;
    const int kb = (lane >> 4) << 3;
    const bf16* p = A + m * lda + kt * 32 + kb;
    v8bf lo = *(const v8bf*)p;
    v8bf hi = *(const v8bf*)(p + 16);
    v16bf r;
#pragma unroll
    for (int i = 0; i < 8; ++i) { r[i] = lo[i]; r[i + 8] = hi[i]; }
    return r;
}

// B fragment (32x16 bf16, KxN) from pre-packed weights: 32 lanes x 16 bf16
// contiguous per tile (1024B/tile, 32B/lane).
__device__ __forceinline__ v16bf gbl_b_frag(const bf16* Bt, int lane) {
    const bf16* p = Bt + lane * 16;
    v8bf lo = *(const v8bf*)p;
    v8bf hi = *(const v8bf*)(p + 8);
    v16bf r;
#pragma unroll
    for (int i = 0; i < 8; ++i) { r[i] = lo[i]; r[i + 8] = hi[i]; }
    return r;
}

__device__ __forceinline__ v8f wmma_bf16(v16bf a, v16bf b, v8f c) {
    return __builtin_amdgcn_wmma_f32_16x16x32_bf16(false, a, false, b, (short)0, c,
                                                   false, false);
}

#define V8F_ZERO {0.f,0.f,0.f,0.f,0.f,0.f,0.f,0.f}

// ------------------------------------------------------------ pack kernel --
// W: f32 [K][128] row-major -> packed bf16 tiles, tile t = nt*(K/32)+kt,
// element (t, lane, j) = W[kt*32 + (lane>>4)*8 + (j&7) + (j>>3)*16][nt*16 + (lane&15)].
__global__ void k_pack_w(const float* __restrict__ W, bf16* __restrict__ out, int K) {
    int tiles_k = K >> 5;
    int total   = 8 * tiles_k * 32;
    int gid = blockIdx.x * blockDim.x + threadIdx.x;
    if (gid >= total) return;
    int lane = gid & 31;
    int t    = gid >> 5;
    int kt   = t % tiles_k;
    int nt   = t / tiles_k;
    int n    = nt * 16 + (lane & 15);
    int kb   = kt * 32 + ((lane >> 4) << 3);
    bf16* o  = out + (size_t)t * 512 + lane * 16;
#pragma unroll
    for (int j = 0; j < 16; ++j) o[j] = (bf16)W[(size_t)(kb + (j & 7) + ((j >> 3) << 4)) * 128 + n];
}

// --------------------------------------------------- fused gated-MLP kernel --
// out[r] = silu(LN(silu(x@cw1)@cw2)) * sigmoid(LN(silu(x@gw1)@gw2)) (+res)
// x = concat of nsrc gathered [.,128] f32 sources. 16 rows/block, 8 waves.
__global__ __launch_bounds__(256) void k_gmlp(
    int nrows,
    const float* __restrict__ s0, const int* __restrict__ i0,
    const float* __restrict__ s1, const int* __restrict__ i1,
    const float* __restrict__ s2, const int* __restrict__ i2,
    const float* __restrict__ s3, const int* __restrict__ i3,
    int nsrc,
    const bf16* __restrict__ cw1p, const bf16* __restrict__ cw2p,
    const bf16* __restrict__ gw1p, const bf16* __restrict__ gw2p,
    const float* __restrict__ cg, const float* __restrict__ cb,
    const float* __restrict__ gg, const float* __restrict__ gb,
    const float* __restrict__ resp, const float* __restrict__ res_scale,
    float* __restrict__ out)
{
    __shared__ bf16  Ax[16 * 512];
    __shared__ bf16  H [16 * 128];
    __shared__ float C2[16 * 128];
    __shared__ float G2[16 * 128];

    const int tid  = threadIdx.x;
    const int din  = nsrc * 128;
    const int row0 = blockIdx.x * 16;
    const int q    = din >> 2;  // float4 per row

    // gather + f32->bf16 convert into LDS
    for (int i = tid; i < 16 * q; i += 256) {
        int r = i / q;
        int c = (i - r * q) * 4;
        int grow = row0 + r; if (grow >= nrows) grow = nrows - 1;
        const float* sp; const int* ip;
        int seg = c >> 7;
        if      (seg == 0) { sp = s0; ip = i0; }
        else if (seg == 1) { sp = s1; ip = i1; }
        else if (seg == 2) { sp = s2; ip = i2; }
        else               { sp = s3; ip = i3; }
        int srow = ip ? ip[grow] : grow;
        float4 v = *(const float4*)(sp + (size_t)srow * 128 + (c & 127));
        bf16* d = Ax + r * din + c;
        d[0] = (bf16)v.x; d[1] = (bf16)v.y; d[2] = (bf16)v.z; d[3] = (bf16)v.w;
    }
    __syncthreads();

    const int wid  = tid >> 5;
    const int lane = tid & 31;
    const int n    = wid * 16 + (lane & 15);
    const int mb   = (lane >> 4) << 3;
    const int tk   = din >> 5;

    // ---- core chain ----
    {
        v8f acc = V8F_ZERO;
        for (int kt = 0; kt < tk; ++kt) {
            const bf16* bt = cw1p + (((size_t)wid * tk + kt) << 9);
            if (kt + 1 < tk) __builtin_prefetch(bt + 512, 0, 1);
            acc = wmma_bf16(lds_a_frag(Ax, din, kt, lane), gbl_b_frag(bt, lane), acc);
        }
#pragma unroll
        for (int i = 0; i < 8; ++i) H[(mb + i) * 128 + n] = (bf16)siluf(acc[i]);
    }
    __syncthreads();
    {
        v8f acc = V8F_ZERO;
        for (int kt = 0; kt < 4; ++kt)
            acc = wmma_bf16(lds_a_frag(H, 128, kt, lane),
                            gbl_b_frag(cw2p + (((size_t)wid * 4 + kt) << 9), lane), acc);
#pragma unroll
        for (int i = 0; i < 8; ++i) C2[(mb + i) * 128 + n] = acc[i];
    }
    __syncthreads();

    // ---- gate chain ----
    {
        v8f acc = V8F_ZERO;
        for (int kt = 0; kt < tk; ++kt) {
            const bf16* bt = gw1p + (((size_t)wid * tk + kt) << 9);
            if (kt + 1 < tk) __builtin_prefetch(bt + 512, 0, 1);
            acc = wmma_bf16(lds_a_frag(Ax, din, kt, lane), gbl_b_frag(bt, lane), acc);
        }
#pragma unroll
        for (int i = 0; i < 8; ++i) H[(mb + i) * 128 + n] = (bf16)siluf(acc[i]);
    }
    __syncthreads();
    {
        v8f acc = V8F_ZERO;
        for (int kt = 0; kt < 4; ++kt)
            acc = wmma_bf16(lds_a_frag(H, 128, kt, lane),
                            gbl_b_frag(gw2p + (((size_t)wid * 4 + kt) << 9), lane), acc);
#pragma unroll
        for (int i = 0; i < 8; ++i) G2[(mb + i) * 128 + n] = acc[i];
    }
    __syncthreads();

    // ---- LayerNorm (per row) + activations + gate + residual ----
    {
        const int r  = tid >> 4;        // 16 threads per row
        const int c0 = (tid & 15) << 3; // 8 columns each
        float sc = 0.f, sc2 = 0.f, sg = 0.f, sg2 = 0.f;
#pragma unroll
        for (int j = 0; j < 8; ++j) {
            float x = C2[r * 128 + c0 + j]; sc += x; sc2 += x * x;
            float y = G2[r * 128 + c0 + j]; sg += y; sg2 += y * y;
        }
#pragma unroll
        for (int o = 8; o >= 1; o >>= 1) {
            sc  += __shfl_xor(sc,  o, 16);
            sc2 += __shfl_xor(sc2, o, 16);
            sg  += __shfl_xor(sg,  o, 16);
            sg2 += __shfl_xor(sg2, o, 16);
        }
        const float mc = sc * (1.f / 128.f);
        const float ic = rsqrtf(fmaxf(sc2 * (1.f / 128.f) - mc * mc, 0.f) + 1e-5f);
        const float mg = sg * (1.f / 128.f);
        const float ig = rsqrtf(fmaxf(sg2 * (1.f / 128.f) - mg * mg, 0.f) + 1e-5f);
        const int grow = row0 + r;
        if (grow < nrows) {
            float* op = out + (size_t)grow * 128 + c0;
            const float* rp = resp ? resp + (size_t)grow * 128 + c0 : nullptr;
#pragma unroll
            for (int j = 0; j < 8; ++j) {
                int c = c0 + j;
                float xc = (C2[r * 128 + c] - mc) * ic * cg[c] + cb[c];
                float xg = (G2[r * 128 + c] - mg) * ig * gg[c] + gb[c];
                float o  = siluf(xc) * sigmf(xg);
                if (rp) o += res_scale[c] * rp[j];
                op[j] = o;
            }
        }
    }
}

// --------------------------------------------------------- plain MLP kernel --
// out = silu(X@W1)@W2 (+ res_scale*res). din = 128, no gather.
__global__ __launch_bounds__(256) void k_mlp(
    int nrows, const float* __restrict__ X,
    const bf16* __restrict__ w1p, const bf16* __restrict__ w2p,
    const float* __restrict__ resp, const float* __restrict__ res_scale,
    float* __restrict__ out)
{
    __shared__ bf16 Ax[16 * 128];
    __shared__ bf16 H [16 * 128];
    const int tid = threadIdx.x, row0 = blockIdx.x * 16;
    for (int i = tid; i < 16 * 32; i += 256) {
        int r = i >> 5, c = (i & 31) * 4;
        int grow = row0 + r; if (grow >= nrows) grow = nrows - 1;
        float4 v = *(const float4*)(X + (size_t)grow * 128 + c);
        bf16* d = Ax + r * 128 + c;
        d[0] = (bf16)v.x; d[1] = (bf16)v.y; d[2] = (bf16)v.z; d[3] = (bf16)v.w;
    }
    __syncthreads();
    const int wid = tid >> 5, lane = tid & 31;
    const int n = wid * 16 + (lane & 15), mb = (lane >> 4) << 3;
    {
        v8f acc = V8F_ZERO;
        for (int kt = 0; kt < 4; ++kt)
            acc = wmma_bf16(lds_a_frag(Ax, 128, kt, lane),
                            gbl_b_frag(w1p + (((size_t)wid * 4 + kt) << 9), lane), acc);
#pragma unroll
        for (int i = 0; i < 8; ++i) H[(mb + i) * 128 + n] = (bf16)siluf(acc[i]);
    }
    __syncthreads();
    {
        v8f acc = V8F_ZERO;
        for (int kt = 0; kt < 4; ++kt)
            acc = wmma_bf16(lds_a_frag(H, 128, kt, lane),
                            gbl_b_frag(w2p + (((size_t)wid * 4 + kt) << 9), lane), acc);
#pragma unroll
        for (int i = 0; i < 8; ++i) {
            int grow = row0 + mb + i;
            if (grow < nrows) {
                float o = acc[i];
                if (resp) o += res_scale[n] * resp[(size_t)grow * 128 + n];
                out[(size_t)grow * 128 + n] = o;
            }
        }
    }
}

// -------------------------------------------------- single GEMM with gather --
// out = gather(X, idx) @ W  (attention logits)
__global__ __launch_bounds__(256) void k_gemm1(
    int nrows, const float* __restrict__ X, const int* __restrict__ idx,
    const bf16* __restrict__ Wp, float* __restrict__ out)
{
    __shared__ bf16 Ax[16 * 128];
    const int tid = threadIdx.x, row0 = blockIdx.x * 16;
    for (int i = tid; i < 16 * 32; i += 256) {
        int r = i >> 5, c = (i & 31) * 4;
        int grow = row0 + r; if (grow >= nrows) grow = nrows - 1;
        int srow = idx ? idx[grow] : grow;
        float4 v = *(const float4*)(X + (size_t)srow * 128 + c);
        bf16* d = Ax + r * 128 + c;
        d[0] = (bf16)v.x; d[1] = (bf16)v.y; d[2] = (bf16)v.z; d[3] = (bf16)v.w;
    }
    __syncthreads();
    const int wid = tid >> 5, lane = tid & 31;
    const int n = wid * 16 + (lane & 15), mb = (lane >> 4) << 3;
    v8f acc = V8F_ZERO;
    for (int kt = 0; kt < 4; ++kt)
        acc = wmma_bf16(lds_a_frag(Ax, 128, kt, lane),
                        gbl_b_frag(Wp + (((size_t)wid * 4 + kt) << 9), lane), acc);
#pragma unroll
    for (int i = 0; i < 8; ++i) {
        int grow = row0 + mb + i;
        if (grow < nrows) out[(size_t)grow * 128 + n] = acc[i];
    }
}

// -------------------------------------------------------- elementwise kernels
#define GS_LOOP(nelts) \
    for (size_t i = (size_t)blockIdx.x * blockDim.x + threadIdx.x; i < (nelts); \
         i += (size_t)gridDim.x * blockDim.x)

__global__ void k_fill(float* p, size_t n, float v) { GS_LOOP(n) p[i] = v; }

__global__ void k_segmax(size_t ne, const float* a, const int* seg, float* m) {
    GS_LOOP(ne * 128) {
        size_t e = i >> 7; int c = i & 127;
        atomicMaxF32(&m[(size_t)seg[e] * 128 + c], a[i]);
    }
}
__global__ void k_expsum(size_t ne, float* a, const int* seg, const float* m, float* s) {
    GS_LOOP(ne * 128) {
        size_t e = i >> 7; int c = i & 127;
        float ex = __expf(a[i] - m[(size_t)seg[e] * 128 + c]);
        a[i] = ex;
        atomicAdd(&s[(size_t)seg[e] * 128 + c], ex);
    }
}
__global__ void k_norm_acc(size_t ne, const float* a, const int* seg, const float* s,
                           const float* ae, float* acc) {
    GS_LOOP(ne * 128) {
        size_t e = i >> 7; int c = i & 127;
        float w = a[i] / s[(size_t)seg[e] * 128 + c];
        atomicAdd(&acc[(size_t)seg[e] * 128 + c], w * ae[i]);
    }
}
__global__ void k_seg_add(size_t ne, const float* x, const int* seg, float* acc, float* cnt) {
    GS_LOOP(ne * 128) {
        size_t e = i >> 7; int c = i & 127;
        atomicAdd(&acc[(size_t)seg[e] * 128 + c], x[i]);
        if (cnt && c == 0) atomicAdd(&cnt[seg[e]], 1.f);
    }
}
// out = acc (/ max(cnt,1)) + res_scale[c]*base
__global__ void k_edge_update(size_t nrows, const float* acc, const float* cnt,
                              const float* base, const float* res_scale, float* out) {
    GS_LOOP(nrows * 128) {
        size_t r = i >> 7; int c = i & 127;
        float v = acc[i];
        if (cnt) v /= fmaxf(cnt[r], 1.f);
        out[i] = v + res_scale[c] * base[i];
    }
}
// out[r,c] = sum_j radial[r,j]*envW[j,c]   (K=7 learnable envelope)
__global__ void k_env(size_t nrows, const float* radial, const float* envW, float* out) {
    GS_LOOP(nrows * 128) {
        size_t r = i >> 7; int c = i & 127;
        float s = 0.f;
#pragma unroll
        for (int j = 0; j < 7; ++j) s += radial[r * 7 + j] * envW[j * 128 + c];
        out[i] = s;
    }
}
// acc[seg] += nl * envb[envidx]
__global__ void k_wsum_env(size_t ne, const float* nl, const float* envb,
                           const int* envidx, const int* seg, float* acc) {
    GS_LOOP(ne * 128) {
        size_t e = i >> 7; int c = i & 127;
        float v = nl[i] * envb[(size_t)envidx[e] * 128 + c];
        atomicAdd(&acc[(size_t)seg[e] * 128 + c], v);
    }
}
// acc[tgt] += nl * envb[src] * envb[tgt]
__global__ void k_wsum_env2(size_t ne, const float* nl, const float* envb,
                            const int* src, const int* tgt, float* acc) {
    GS_LOOP(ne * 128) {
        size_t e = i >> 7; int c = i & 127;
        float v = nl[i] * envb[(size_t)src[e] * 128 + c] * envb[(size_t)tgt[e] * 128 + c];
        atomicAdd(&acc[(size_t)tgt[e] * 128 + c], v);
    }
}

// ------------------------------------------------------------------- host --

extern "C" void kernel_launch(void* const* d_in, const int* in_sizes, int n_in,
                              void* d_out, int out_size, void* d_ws, size_t ws_size,
                              hipStream_t stream) {
    (void)in_sizes; (void)n_in; (void)out_size; (void)ws_size;

    const float* atom_in  = (const float*)d_in[0];
    const float* bond_in  = (const float*)d_in[1];
    const float* angle_in = (const float*)d_in[2];
    const float* radial   = (const float*)d_in[3];
    // params flattened jax-pytree style (dict keys sorted): see mapping below.
    auto P = [&](int i) { return (const float*)d_in[i]; };
    const int* ag_src  = (const int*)d_in[74];
    const int* ag_tgt  = (const int*)d_in[75];
    const int* lg_src  = (const int*)d_in[76];
    const int* lg_tgt  = (const int*)d_in[77];
    const int* lg_atom = (const int*)d_in[78];
    const int* d2u     = (const int*)d_in[79];

    float* out_atom  = (float*)d_out;
    float* out_bond  = out_atom + (size_t)NA_N * 128;
    float* out_angle = out_bond + (size_t)NB_N * 128;

    // bump allocator over workspace
    char* wsb = (char*)d_ws;
    size_t off = 0;
    auto alloc = [&](size_t bytes) -> void* {
        void* p = wsb + off;
        off = (off + bytes + 255) & ~(size_t)255;
        return p;
    };

    bf16* WP = (bf16*)alloc((size_t)2200000 * sizeof(bf16));
    size_t woff = 0;
    auto pack = [&](int pidx, int K) -> const bf16* {
        bf16* dst = WP + woff;
        woff += (size_t)K * 128;
        int total = 8 * (K / 32) * 32;
        k_pack_w<<<(total + 255) / 256, 256, 0, stream>>>(P(pidx), dst, K);
        return dst;
    };

    // pack every weight matrix (indices per sorted-key pytree flatten)
    const bf16 *gaWs = pack(4, 128),  *gaWt = pack(5, 128);
    const bf16 *gaEc1 = pack(8, 384), *gaEc2 = pack(9, 128),
               *gaEg1 = pack(12, 384), *gaEg2 = pack(13, 128);
    const bf16 *gaNc1 = pack(17, 384), *gaNc2 = pack(18, 128),
               *gaNg1 = pack(21, 384), *gaNg2 = pack(22, 128);
    const bf16 *laWs = pack(24, 128), *laWt = pack(25, 128);
    const bf16 *laEc1 = pack(28, 384), *laEc2 = pack(29, 128),
               *laEg1 = pack(32, 384), *laEg2 = pack(33, 128);
    const bf16 *laNc1 = pack(37, 384), *laNc2 = pack(38, 128),
               *laNg1 = pack(41, 384), *laNg2 = pack(42, 128);
    const bf16 *raF1 = pack(45, 128), *raF2 = pack(46, 128);
    const bf16 *raGc1 = pack(50, 384), *raGc2 = pack(51, 128),
               *raGg1 = pack(54, 384), *raGg2 = pack(55, 128);
    const bf16 *raN1 = pack(56, 128), *raN2 = pack(57, 128);
    const bf16 *rlF1 = pack(60, 128), *rlF2 = pack(61, 128);
    const bf16 *rlGc1 = pack(65, 512), *rlGc2 = pack(66, 128),
               *rlGg1 = pack(69, 512), *rlGg2 = pack(70, 128);
    const bf16 *rlN1 = pack(71, 128), *rlN2 = pack(72, 128);

    float* EDa  = (float*)alloc((size_t)ED_N * 128 * 4);
    float* EDb  = (float*)alloc((size_t)ED_N * 128 * 4);
    float* NBa  = (float*)alloc((size_t)NB_N * 128 * 4);
    float* NBb  = (float*)alloc((size_t)NB_N * 128 * 4);
    float* NBc  = (float*)alloc((size_t)NB_N * 128 * 4);
    float* NBd  = (float*)alloc((size_t)NB_N * 128 * 4);
    float* SEGm = (float*)alloc((size_t)NB_N * 128 * 4);
    float* SEGs = (float*)alloc((size_t)NB_N * 128 * 4);
    float* NAa  = (float*)alloc((size_t)NA_N * 128 * 4);
    float* NAb  = (float*)alloc((size_t)NA_N * 128 * 4);
    float* ANG1 = (float*)alloc((size_t)NL_N * 128 * 4);
    float* CNT  = (float*)alloc((size_t)NB_N * 4);

    const int gED = (ED_N + 15) / 16, gNA = (NA_N + 15) / 16;
    const int gNB = (NB_N + 15) / 16, gNL = (NL_N + 15) / 16;
    dim3 B(256);
    auto gfor = [&](size_t n) {
        size_t b = (n + 255) / 256;
        if (b > 262144) b = 262144;
        return dim3((unsigned)b);
    };
    auto fill = [&](float* p, size_t n, float v) {
        k_fill<<<gfor(n), B, 0, stream>>>(p, n, v);
    };

    // ================= stage 1: atom-graph attention (ga) =================
    // attn_e = gmlp([bond[d2u], atom[tgt], atom[src]])
    k_gmlp<<<gED, B, 0, stream>>>(ED_N, bond_in, d2u, atom_in, ag_tgt, atom_in, ag_src,
                                  nullptr, nullptr, 3, gaEc1, gaEc2, gaEg1, gaEg2,
                                  P(7), P(6), P(11), P(10), nullptr, nullptr, EDa);
    // source softmax -> asf
    k_gemm1<<<gED, B, 0, stream>>>(ED_N, bond_in, d2u, gaWs, EDb);
    fill(SEGm, (size_t)NA_N * 128, -1e30f);
    fill(SEGs, (size_t)NA_N * 128, 0.f);
    fill(NAa, (size_t)NA_N * 128, 0.f);
    k_segmax<<<gfor((size_t)ED_N * 128), B, 0, stream>>>(ED_N, EDb, ag_src, SEGm);
    k_expsum<<<gfor((size_t)ED_N * 128), B, 0, stream>>>(ED_N, EDb, ag_src, SEGm, SEGs);
    k_norm_acc<<<gfor((size_t)ED_N * 128), B, 0, stream>>>(ED_N, EDb, ag_src, SEGs, EDa, NAa);
    // target softmax -> atf
    k_gemm1<<<gED, B, 0, stream>>>(ED_N, bond_in, d2u, gaWt, EDb);
    fill(SEGm, (size_t)NA_N * 128, -1e30f);
    fill(SEGs, (size_t)NA_N * 128, 0.f);
    fill(NAb, (size_t)NA_N * 128, 0.f);
    k_segmax<<<gfor((size_t)ED_N * 128), B, 0, stream>>>(ED_N, EDb, ag_tgt, SEGm);
    k_expsum<<<gfor((size_t)ED_N * 128), B, 0, stream>>>(ED_N, EDb, ag_tgt, SEGm, SEGs);
    k_norm_acc<<<gfor((size_t)ED_N * 128), B, 0, stream>>>(ED_N, EDb, ag_tgt, SEGs, EDa, NAb);
    // attn_e mean over d2u -> bond_v1
    fill(NBa, (size_t)NB_N * 128, 0.f);
    fill(CNT, NB_N, 0.f);
    k_seg_add<<<gfor((size_t)ED_N * 128), B, 0, stream>>>(ED_N, EDa, d2u, NBa, CNT);
    k_edge_update<<<gfor((size_t)NB_N * 128), B, 0, stream>>>(NB_N, NBa, CNT, bond_in,
                                                              P(14), NBb);  // bond_v1
    // atom_v1 = gmlp([atom, atf, asf]) + node_res*atom
    k_gmlp<<<gNA, B, 0, stream>>>(NA_N, atom_in, nullptr, NAb, nullptr, NAa, nullptr,
                                  nullptr, nullptr, 3, gaNc1, gaNc2, gaNg1, gaNg2,
                                  P(16), P(15), P(20), P(19), atom_in, P(23), out_atom);

    // ================= stage 2: line-graph attention (la) =================
    k_gmlp<<<gNL, B, 0, stream>>>(NL_N, angle_in, nullptr, NBb, lg_tgt, NBb, lg_src,
                                  nullptr, nullptr, 3, laEc1, laEc2, laEg1, laEg2,
                                  P(27), P(26), P(31), P(30), nullptr, nullptr, EDa);
    k_gemm1<<<gNL, B, 0, stream>>>(NL_N, angle_in, nullptr, laWs, EDb);
    fill(SEGm, (size_t)NB_N * 128, -1e30f);
    fill(SEGs, (size_t)NB_N * 128, 0.f);
    fill(NBc, (size_t)NB_N * 128, 0.f);
    k_segmax<<<gfor((size_t)NL_N * 128), B, 0, stream>>>(NL_N, EDb, lg_src, SEGm);
    k_expsum<<<gfor((size_t)NL_N * 128), B, 0, stream>>>(NL_N, EDb, lg_src, SEGm, SEGs);
    k_norm_acc<<<gfor((size_t)NL_N * 128), B, 0, stream>>>(NL_N, EDb, lg_src, SEGs, EDa, NBc);
    k_gemm1<<<gNL, B, 0, stream>>>(NL_N, angle_in, nullptr, laWt, EDb);
    fill(SEGm, (size_t)NB_N * 128, -1e30f);
    fill(SEGs, (size_t)NB_N * 128, 0.f);
    fill(NBd, (size_t)NB_N * 128, 0.f);
    k_segmax<<<gfor((size_t)NL_N * 128), B, 0, stream>>>(NL_N, EDb, lg_tgt, SEGm);
    k_expsum<<<gfor((size_t)NL_N * 128), B, 0, stream>>>(NL_N, EDb, lg_tgt, SEGm, SEGs);
    k_norm_acc<<<gfor((size_t)NL_N * 128), B, 0, stream>>>(NL_N, EDb, lg_tgt, SEGs, EDa, NBd);
    // bond_v2 = gmlp([bond_v1, atf_b, asf_b]) + node_res*bond_v1
    k_gmlp<<<gNB, B, 0, stream>>>(NB_N, NBb, nullptr, NBd, nullptr, NBc, nullptr,
                                  nullptr, nullptr, 3, laNc1, laNc2, laNg1, laNg2,
                                  P(36), P(35), P(40), P(39), NBb, P(43), NBa);
    // angle_v1 = attn_e_l + edge_res*angle
    k_edge_update<<<gfor((size_t)NL_N * 128), B, 0, stream>>>(NL_N, EDa, nullptr,
                                                              angle_in, P(34), ANG1);

    // ================= stage 3: refine_atom (ra) =================
    k_env<<<gfor((size_t)NB_N * 128), B, 0, stream>>>(NB_N, radial, P(47), NBc);
    k_gmlp<<<gED, B, 0, stream>>>(ED_N, NBa, d2u, out_atom, ag_tgt, out_atom, ag_src,
                                  nullptr, nullptr, 3, raGc1, raGc2, raGg1, raGg2,
                                  P(49), P(48), P(53), P(52), nullptr, nullptr, EDa);
    fill(NAa, (size_t)NA_N * 128, 0.f);
    k_wsum_env<<<gfor((size_t)ED_N * 128), B, 0, stream>>>(ED_N, EDa, NBc, d2u, ag_tgt, NAa);
    // atom_v2 = mlp(rn) + node_res*atom_v1  (in-place on out_atom)
    k_mlp<<<gNA, B, 0, stream>>>(NA_N, NAa, raN1, raN2, out_atom, P(58), out_atom);
    // de = mlp(nl); mean over d2u; bond_v3 = mean + edge_res*bond_v2
    k_mlp<<<gED, B, 0, stream>>>(ED_N, EDa, raF1, raF2, nullptr, nullptr, EDb);
    fill(NBd, (size_t)NB_N * 128, 0.f);
    k_seg_add<<<gfor((size_t)ED_N * 128), B, 0, stream>>>(ED_N, EDb, d2u, NBd, nullptr);
    k_edge_update<<<gfor((size_t)NB_N * 128), B, 0, stream>>>(NB_N, NBd, CNT, NBa,
                                                              P(44), NBb);  // bond_v3

    // ================= stage 4: refine_line (rl) =================
    k_env<<<gfor((size_t)NB_N * 128), B, 0, stream>>>(NB_N, radial, P(62), NBc);
    k_gmlp<<<gNL, B, 0, stream>>>(NL_N, ANG1, nullptr, out_atom, lg_atom, NBb, lg_tgt,
                                  NBb, lg_src, 4, rlGc1, rlGc2, rlGg1, rlGg2,
                                  P(64), P(63), P(68), P(67), nullptr, nullptr, EDa);
    fill(NBd, (size_t)NB_N * 128, 0.f);
    k_wsum_env2<<<gfor((size_t)NL_N * 128), B, 0, stream>>>(NL_N, EDa, NBc, lg_src,
                                                            lg_tgt, NBd);
    // final bond = mlp(rn_b) + node_res*bond_v3
    k_mlp<<<gNB, B, 0, stream>>>(NB_N, NBd, rlN1, rlN2, NBb, P(73), out_bond);
    // final angle = mlp(nl_l) + edge_res*angle_v1
    k_mlp<<<gNL, B, 0, stream>>>(NL_N, EDa, rlF1, rlF2, ANG1, P(59), out_angle);
}
